// MatchingHead_46961172415014
// MI455X (gfx1250) — matrix operand order repaired and checked by reading, not compiled, not run on hardware
//
#include <hip/hip_runtime.h>
#include <hip/hip_bf16.h>
#include <math.h>

typedef __attribute__((ext_vector_type(16))) _Float16 v16h;
typedef __attribute__((ext_vector_type(8)))  _Float16 v8h;
typedef __attribute__((ext_vector_type(8)))  float    v8f;
typedef __attribute__((ext_vector_type(4)))  unsigned v4u;
typedef __attribute__((ext_vector_type(8)))  int      v8i_;
typedef __attribute__((ext_vector_type(4)))  int      v4i_;

#define Dm   128
#define NTOK 256

// ---------------------------------------------------------------------------
// TDM helper: 2-D f16 tile load (tile_w x tile_h elems, row stride in elems)
// Descriptor per CDNA5 ISA §8.3/8.4: count=1, data_size=1(2B), type=2.
// This toolchain exposes the 6-arg builtin: (g0, g1, g2, g3, g4, cpol).
// ---------------------------------------------------------------------------
__device__ __forceinline__ void tdm_load_2d_f16(unsigned lds_off,
                                                const _Float16* gptr,
                                                int tile_w, int tile_h,
                                                int stride_elems) {
#if __has_builtin(__builtin_amdgcn_tensor_load_to_lds)
  unsigned long long ga = (unsigned long long)(uintptr_t)gptr;
  v4u g0;
  g0[0] = 1u;                                   // count=1, user mode
  g0[1] = lds_off;                              // lds_addr (bytes)
  g0[2] = (unsigned)ga;                         // global_addr[31:0]
  g0[3] = ((unsigned)(ga >> 32) & 0x01FFFFFFu)  // global_addr[56:32]
        | (2u << 30);                           // type=2 ("image")
  v8i_ g1;
  g1[0] = 0x00010000;                           // data_size=1 (2 bytes)
  g1[1] = (tile_w & 0xFFFF) << 16;              // tensor_dim0[15:0]
  g1[2] = ((tile_w >> 16) & 0xFFFF) | ((tile_h & 0xFFFF) << 16);  // d0 hi | d1 lo
  g1[3] = ((tile_h >> 16) & 0xFFFF) | ((tile_w & 0xFFFF) << 16);  // d1 hi | tile_dim0
  g1[4] = (tile_h & 0xFFFF);                    // tile_dim1 (tile_dim2=0)
  g1[5] = stride_elems;                         // tensor_dim0_stride lo32
  g1[6] = 0;
  g1[7] = 0;
  v4i_ z4 = {0, 0, 0, 0};
  v8i_ z8 = {0, 0, 0, 0, 0, 0, 0, 0};
  __builtin_amdgcn_tensor_load_to_lds(g0, g1, z4, z4, z8, 0);
#else
  (void)lds_off; (void)gptr; (void)tile_w; (void)tile_h; (void)stride_elems;
#endif
}

// ---------------------------------------------------------------------------
// Weight transpose + f16 convert: Wt[n*K + k] = (f16)W[k*N + n]
// ---------------------------------------------------------------------------
__global__ void k_wt(const float* __restrict__ W, int K, int N,
                     _Float16* __restrict__ Wt, int total) {
  int i = blockIdx.x * 256 + threadIdx.x;
  if (i < total) {
    int k = i / N, n = i % N;
    Wt[(size_t)n * K + k] = (_Float16)W[i];
  }
}

// ---------------------------------------------------------------------------
// Encode kernel (one 128-thread block per token)
// ---------------------------------------------------------------------------
__global__ __launch_bounds__(128) void k_encode(
    const float* __restrict__ backbone, const float* __restrict__ coords,
    int t, int type_id,
    const float* __restrict__ cw, const float* __restrict__ cb,
    const float* __restrict__ rw, const float* __restrict__ rb,
    const float* __restrict__ rlg, const float* __restrict__ rlb,
    const float* __restrict__ fw, const float* __restrict__ fb,
    const float* __restrict__ flg, const float* __restrict__ flb,
    const float* __restrict__ temb, float* __restrict__ out) {
  const int H = 64, W = 64, CB = 256, T = 2;
  __shared__ float four[24];
  __shared__ float samp[256];
  __shared__ float z[128];
  __shared__ float roi[64];
  __shared__ float mv[2];
  int b = blockIdx.x >> 8, n = blockIdx.x & 255, tid = threadIdx.x;
  float cx = coords[(b * NTOK + n) * 2 + 0];
  float cy = coords[(b * NTOK + n) * 2 + 1];
  if (tid < 24) {
    int comp = tid / 6, fi = tid % 6;
    float f = (float)(1 << fi) * 3.14159265358979323846f;
    float base = (comp == 0 || comp == 2) ? cx : cy;
    four[tid] = (comp < 2) ? sinf(base * f) : cosf(base * f);
  }
  float ix = cx * (float)(W - 1), iy = cy * (float)(H - 1);
  float x0f = floorf(ix), y0f = floorf(iy);
  float wx = ix - x0f, wy = iy - y0f;
  int x0 = (int)x0f; if (x0 < 0) x0 = 0; if (x0 > W - 1) x0 = W - 1;
  int y0 = (int)y0f; if (y0 < 0) y0 = 0; if (y0 > H - 1) y0 = H - 1;
  int x1 = x0 + 1; if (x1 > W - 1) x1 = W - 1;
  int y1 = y0 + 1; if (y1 > H - 1) y1 = H - 1;
  // prefetch second channel's cacheline while first is processed
  {
    const float* fmp = backbone + (((size_t)b * CB + tid + 128) * T + t) * (size_t)(H * W);
    __builtin_prefetch(fmp + y0 * W + x0, 0, 1);
  }
#pragma unroll
  for (int cc = 0; cc < 2; ++cc) {
    int ch = tid + cc * 128;
    const float* fm = backbone + (((size_t)b * CB + ch) * T + t) * (size_t)(H * W);
    float v00 = fm[y0 * W + x0], v01 = fm[y0 * W + x1];
    float v10 = fm[y1 * W + x0], v11 = fm[y1 * W + x1];
    samp[ch] = v00 * (1.f - wx) * (1.f - wy) + v01 * wx * (1.f - wy)
             + v10 * (1.f - wx) * wy + v11 * wx * wy;
  }
  __syncthreads();
  if (tid < 64) {
    float a = cb[tid];
    for (int j = 0; j < 24; ++j) a += four[j] * cw[j * 64 + tid];
    z[tid] = fmaxf(a, 0.f);
    float r = rb[tid];
    for (int j = 0; j < 256; ++j) r += samp[j] * rw[j * 64 + tid];
    roi[tid] = fmaxf(r, 0.f);
  }
  __syncthreads();
  if (tid == 0) {
    float m = 0.f; for (int i = 0; i < 64; ++i) m += roi[i]; m *= (1.f / 64.f);
    float v = 0.f; for (int i = 0; i < 64; ++i) { float d = roi[i] - m; v += d * d; }
    v *= (1.f / 64.f);
    mv[0] = m; mv[1] = rsqrtf(v + 1e-5f);
  }
  __syncthreads();
  if (tid < 64) z[64 + tid] = (roi[tid] - mv[0]) * mv[1] * rlg[tid] + rlb[tid];
  __syncthreads();
  float fz = fb[tid];
  for (int j = 0; j < 128; ++j) fz += z[j] * fw[j * 128 + tid];
  fz = fmaxf(fz, 0.f);
  samp[tid] = fz;
  __syncthreads();
  if (tid == 0) {
    float m = 0.f; for (int i = 0; i < 128; ++i) m += samp[i]; m *= (1.f / 128.f);
    float v = 0.f; for (int i = 0; i < 128; ++i) { float d = samp[i] - m; v += d * d; }
    v *= (1.f / 128.f);
    mv[0] = m; mv[1] = rsqrtf(v + 1e-5f);
  }
  __syncthreads();
  out[((size_t)b * NTOK + n) * Dm + tid] =
      (fz - mv[0]) * mv[1] * flg[tid] + flb[tid] + temb[type_id * Dm + tid];
}

// ---------------------------------------------------------------------------
// WMMA GEMM: C = A[M,K](f32) @ Wt^T (Wt is f16 [N][K]) (+bias)(+relu)
// Block = 8 waves -> 16 rows x 128 cols; grid = (M/16, N/128).
// B fragment = one contiguous 32B load; A fragment = two 16B LDS loads.
// Optional f16 secondary output C16.
// ---------------------------------------------------------------------------
__global__ __launch_bounds__(256) void k_gemm(
    const float* __restrict__ A, int lda,
    const _Float16* __restrict__ Wt,
    const float* __restrict__ bias,
    float* __restrict__ C, int ldc,
    _Float16* __restrict__ C16, int ldc16,
    int K, int relu) {
  __shared__ __align__(32) _Float16 Ash[16 * 32];
  const int tid = threadIdx.x;
  const int wave = tid >> 5;
  const int lane = tid & 31;
  const int m0 = blockIdx.x * 16;
  const int ncol = blockIdx.y * 128 + wave * 16 + (lane & 15);
  const int khalf = (lane >> 4) << 4;
  const int kbB   = (lane >> 4) << 4;
  const int mrow  = lane & 15;
  const _Float16* wrow = Wt + (size_t)ncol * K;
  v8f acc = {};
  for (int kt = 0; kt < K; kt += 32) {
    int e = tid * 2;
    int r = e >> 5, c = e & 31;
    Ash[r * 32 + c]     = (_Float16)A[(m0 + r) * lda + kt + c];
    Ash[r * 32 + c + 1] = (_Float16)A[(m0 + r) * lda + kt + c + 1];
    __syncthreads();
    v8h alo = *(const v8h*)&Ash[mrow * 32 + khalf];
    v8h ahi = *(const v8h*)&Ash[mrow * 32 + 8 + khalf];
    v16h af = __builtin_shufflevector(alo, ahi, 0, 1, 2, 3, 4, 5, 6, 7,
                                      8, 9, 10, 11, 12, 13, 14, 15);
    v16h bf = *(const v16h*)(wrow + kt + kbB);
    acc = __builtin_amdgcn_wmma_f32_16x16x32_f16(false, af, false, bf,
                                                 (short)0, acc, false, false);
    __syncthreads();
  }
  const int mbase = m0 + ((lane >> 4) << 3);
  float bval = bias ? bias[ncol] : 0.0f;
#pragma unroll
  for (int r = 0; r < 8; ++r) {
    float v = acc[r] + bval;
    if (relu) v = fmaxf(v, 0.0f);
    C[(size_t)(mbase + r) * ldc + ncol] = v;
    if (C16) C16[(size_t)(mbase + r) * ldc16 + ncol] = (_Float16)v;
  }
}

// ---------------------------------------------------------------------------
// Attention: one block per (b, head). K/V tiles (32x256 f16, stride ldkv)
// staged into LDS by the Tensor Data Mover; two-pass online softmax.
// ---------------------------------------------------------------------------
__global__ __launch_bounds__(256) void k_attn(
    const _Float16* __restrict__ q16, int ldq, int qcol0,
    const _Float16* __restrict__ kv16, int ldkv, int kcol0, int vcol0,
    float* __restrict__ out, int Nk) {
  __shared__ __align__(32) _Float16 Ksh[256][32];
  __shared__ __align__(32) _Float16 Vsh[256][32];
  int b = blockIdx.x >> 2;
  int h = blockIdx.x & 3;
  int q = threadIdx.x;
#if __has_builtin(__builtin_amdgcn_tensor_load_to_lds)
  if (threadIdx.x < 32) {   // one wave issues the TDM descriptors
    const _Float16* kbase = kv16 + (size_t)b * Nk * ldkv + kcol0 + h * 32;
    const _Float16* vbase = kv16 + (size_t)b * Nk * ldkv + vcol0 + h * 32;
    tdm_load_2d_f16((unsigned)(uintptr_t)&Ksh[0][0], kbase, 32, Nk, ldkv);
    tdm_load_2d_f16((unsigned)(uintptr_t)&Vsh[0][0], vbase, 32, Nk, ldkv);
    __builtin_amdgcn_s_wait_tensorcnt(0);
  }
#else
  {
    int k = threadIdx.x;
    const _Float16* kr = kv16 + ((size_t)b * Nk + k) * ldkv + kcol0 + h * 32;
    const _Float16* vr = kv16 + ((size_t)b * Nk + k) * ldkv + vcol0 + h * 32;
#pragma unroll
    for (int d = 0; d < 32; ++d) { Ksh[k][d] = kr[d]; Vsh[k][d] = vr[d]; }
  }
#endif
  float qreg[32];
  const _Float16* qr = q16 + ((size_t)b * NTOK + q) * ldq + qcol0 + h * 32;
#pragma unroll
  for (int d = 0; d < 32; ++d) qreg[d] = (float)qr[d];
  __syncthreads();
  const float scale = 0.17677669529663687f;  // 1/sqrt(32)
  float mx = -1e30f, s = 0.f;
  for (int k = 0; k < Nk; ++k) {
    float sc = 0.f;
#pragma unroll
    for (int d = 0; d < 32; ++d) sc += qreg[d] * (float)Ksh[k][d];
    sc *= scale;
    float nm = fmaxf(mx, sc);
    s = s * expf(mx - nm) + expf(sc - nm);
    mx = nm;
  }
  float accv[32];
#pragma unroll
  for (int d = 0; d < 32; ++d) accv[d] = 0.f;
  float inv = 1.f / s;
  for (int k = 0; k < Nk; ++k) {
    float sc = 0.f;
#pragma unroll
    for (int d = 0; d < 32; ++d) sc += qreg[d] * (float)Ksh[k][d];
    float w = expf(sc * scale - mx) * inv;
#pragma unroll
    for (int d = 0; d < 32; ++d) accv[d] += w * (float)Vsh[k][d];
  }
  float* o = out + ((size_t)b * NTOK + q) * Dm + h * 32;
#pragma unroll
  for (int d = 0; d < 32; ++d) o[d] = accv[d];
}

// x = LN(x + sub) * g + b
__global__ __launch_bounds__(128) void k_resln(
    float* __restrict__ x, const float* __restrict__ sub,
    const float* __restrict__ g, const float* __restrict__ bta) {
  __shared__ float sh[128];
  __shared__ float mv[2];
  int row = blockIdx.x, t = threadIdx.x;
  float v = x[(size_t)row * 128 + t] + sub[(size_t)row * 128 + t];
  sh[t] = v;
  __syncthreads();
  if (t == 0) {
    float m = 0.f; for (int i = 0; i < 128; ++i) m += sh[i]; m *= (1.f / 128.f);
    float var = 0.f; for (int i = 0; i < 128; ++i) { float d = sh[i] - m; var += d * d; }
    var *= (1.f / 128.f);
    mv[0] = m; mv[1] = rsqrtf(var + 1e-5f);
  }
  __syncthreads();
  x[(size_t)row * 128 + t] = (v - mv[0]) * mv[1] * g[t] + bta[t];
}

__global__ void k_fill(float* __restrict__ p, int n, float v) {
  int i = blockIdx.x * 256 + threadIdx.x;
  if (i < n) p[i] = v;
}

// ---------------------------------------------------------------------------
// Fused pairwise scorer. Block = (b, p, 16-c tile), 8 waves cover SH=128.
// WaT is f16 [SH][D] (transposed) -> contiguous 32B B-fragment loads.
// Cross-lane shfl_xor reduction replaces shared atomics.
// ---------------------------------------------------------------------------
__global__ __launch_bounds__(256) void k_scorer(
    const float* __restrict__ hs, const float* __restrict__ prev_tok,
    const float* __restrict__ prevWp, const float* __restrict__ hsWc,
    const _Float16* __restrict__ WaT,
    const float* __restrict__ s1_w, const float* __restrict__ s1_b,
    const float* __restrict__ s2_w, const float* __restrict__ s2_b,
    const float* __restrict__ prev_info, const float* __restrict__ curr_info,
    float* __restrict__ aug) {
  __shared__ __align__(32) _Float16 Ash[16 * 128];
  __shared__ float part[8][16];
  __shared__ float cxy[16][2];
  int tid = threadIdx.x, wave = tid >> 5, lane = tid & 31;
  int blk = blockIdx.x;
  int ct = blk & 15;
  int p  = (blk >> 4) & 255;
  int b  = blk >> 12;
  int c0 = ct * 16;
  for (int e = tid; e < 16 * 128; e += 256) {
    int r = e >> 7, c = e & 127;
    float d = hs[((size_t)b * NTOK + c0 + r) * Dm + c]
            - prev_tok[((size_t)b * NTOK + p) * Dm + c];
    Ash[r * 128 + c] = (_Float16)fabsf(d);
  }
  if (tid < 16) {
    cxy[tid][0] = curr_info[((size_t)b * NTOK + c0 + tid) * 2 + 0];
    cxy[tid][1] = curr_info[((size_t)b * NTOK + c0 + tid) * 2 + 1];
  }
  __syncthreads();
  float px = prev_info[((size_t)b * NTOK + p) * 2 + 0];
  float py = prev_info[((size_t)b * NTOK + p) * 2 + 1];
  int n = wave * 16 + (lane & 15);
  int khalf = (lane >> 4) << 4;
  int kbB   = (lane >> 4) << 4;
  int mrow  = lane & 15;
  const _Float16* wrow = WaT + (size_t)n * 128;
  v8f acc = {};
#pragma unroll
  for (int kt = 0; kt < 128; kt += 32) {
    v8h alo = *(const v8h*)&Ash[mrow * 128 + kt + khalf];
    v8h ahi = *(const v8h*)&Ash[mrow * 128 + kt + 8 + khalf];
    v16h af = __builtin_shufflevector(alo, ahi, 0, 1, 2, 3, 4, 5, 6, 7,
                                      8, 9, 10, 11, 12, 13, 14, 15);
    v16h bf = *(const v16h*)(wrow + kt + kbB);
    acc = __builtin_amdgcn_wmma_f32_16x16x32_f16(false, af, false, bf,
                                                 (short)0, acc, false, false);
  }
  float pW  = prevWp[((size_t)b * NTOK + p) * 128 + n];
  float sb  = s1_b[n];
  float wd0 = s1_w[384 * 128 + n], wd1 = s1_w[385 * 128 + n];
  float s2  = s2_w[n];
  int mhalf = (lane >> 4) << 3;
  float pv[8];
#pragma unroll
  for (int r = 0; r < 8; ++r) {
    int m = mhalf + r;
    int c = c0 + m;
    float hW = hsWc[((size_t)b * NTOK + c) * 128 + n];
    float dx = cxy[m][0] - px, dy = cxy[m][1] - py;
    float v = acc[r] + pW + hW + sb + dx * wd0 + dy * wd1;
    v = fmaxf(v, 0.0f);
    pv[r] = v * s2;
  }
  // reduce over the 16 n-lanes within each half-wave
#pragma unroll
  for (int r = 0; r < 8; ++r) {
    float tsum = pv[r];
#pragma unroll
    for (int off = 8; off >= 1; off >>= 1) tsum += __shfl_xor(tsum, off, 16);
    pv[r] = tsum;
  }
  if ((lane & 15) == 0) {
#pragma unroll
    for (int r = 0; r < 8; ++r) part[wave][mhalf + r] = pv[r];
  }
  __syncthreads();
  if (tid < 16) {
    float sum = 0.f;
#pragma unroll
    for (int w = 0; w < 8; ++w) sum += part[w][tid];
    aug[(size_t)b * 257 * 257 + (size_t)p * 257 + (c0 + tid)] = sum + s2_b[0];
  }
}

// dustbin rows/cols
__global__ __launch_bounds__(256) void k_db(
    const float* __restrict__ tok, const float* __restrict__ w,
    const float* __restrict__ bias, float* __restrict__ aug, int isrow) {
  int b = blockIdx.x, i = threadIdx.x;
  float a = bias[0];
  for (int j = 0; j < 128; ++j) a += tok[((size_t)b * NTOK + i) * 128 + j] * w[j];
  if (isrow) aug[(size_t)b * 257 * 257 + (size_t)i * 257 + 256] = a;
  else       aug[(size_t)b * 257 * 257 + 256 * 257 + i] = a;
}

// ---------------------------------------------------------------------------
extern "C" void kernel_launch(void* const* d_in, const int* in_sizes, int n_in,
                              void* d_out, int out_size, void* d_ws, size_t ws_size,
                              hipStream_t stream) {
  (void)in_sizes; (void)n_in; (void)out_size; (void)ws_size;
  const float* backbone  = (const float*)d_in[0];
  const float* prev_info = (const float*)d_in[1];
  const float* curr_info = (const float*)d_in[2];
  const float* coord_w = (const float*)d_in[3];
  const float* coord_b = (const float*)d_in[4];
  const float* roi_w   = (const float*)d_in[5];
  const float* roi_b   = (const float*)d_in[6];
  const float* roi_ln_g = (const float*)d_in[7];
  const float* roi_ln_b = (const float*)d_in[8];
  const float* fuse_w  = (const float*)d_in[9];
  const float* fuse_b  = (const float*)d_in[10];
  const float* fuse_ln_g = (const float*)d_in[11];
  const float* fuse_ln_b = (const float*)d_in[12];
  const float* type_emb  = (const float*)d_in[13];
  const float* sa_w  = (const float*)d_in[14];
  const float* sa_b  = (const float*)d_in[15];
  const float* sa_ow = (const float*)d_in[16];
  const float* sa_ob = (const float*)d_in[17];
  const float* ca_w  = (const float*)d_in[18];
  const float* ca_b  = (const float*)d_in[19];
  const float* ca_ow = (const float*)d_in[20];
  const float* ca_ob = (const float*)d_in[21];
  const float* ff1_w = (const float*)d_in[22];
  const float* ff1_b = (const float*)d_in[23];
  const float* ff2_w = (const float*)d_in[24];
  const float* ff2_b = (const float*)d_in[25];
  const float* ln1_g = (const float*)d_in[26];
  const float* ln1_b = (const float*)d_in[27];
  const float* ln2_g = (const float*)d_in[28];
  const float* ln2_b = (const float*)d_in[29];
  const float* ln3_g = (const float*)d_in[30];
  const float* ln3_b = (const float*)d_in[31];
  const float* s1_w  = (const float*)d_in[32];
  const float* s1_b  = (const float*)d_in[33];
  const float* s2_w  = (const float*)d_in[34];
  const float* s2_b  = (const float*)d_in[35];
  const float* rdb_w = (const float*)d_in[36];
  const float* rdb_b = (const float*)d_in[37];
  const float* cdb_w = (const float*)d_in[38];
  const float* cdb_b = (const float*)d_in[39];

  float* ws = (float*)d_ws;
  float* prev_tok = ws;                 // 131072
  float* x        = ws + 131072;        // 131072
  float* qkv      = ws + 262144;        // 1024*384
  float* kvb      = ws + 655360;        // 1024*384
  float* attn     = ws + 1048576;       // 131072
  float* sub      = ws + 1179648;       // 131072
  float* ffb      = ws + 1310720;       // 1024*2048
  float* prevWp   = ws + 3407872;       // 131072
  float* hsWc     = ws + 3538944;       // 131072
  _Float16* h     = (_Float16*)(ws + 3670016);
  _Float16* qkv16 = h;                  // 1024*384 halves
  _Float16* kv16  = h + 393216;         // 1024*384 halves
  _Float16* wT    = h + 786432;         // transposed f16 weights, 655360/layer
  _Float16* WpT   = h + 786432 + 1310720;
  _Float16* WcT   = WpT + 16384;
  _Float16* WaT   = WcT + 16384;
  float* aug = (float*)d_out;

  // ---- one-time weight transpose+convert (runs on stream, deterministic) ----
  for (int l = 0; l < 2; ++l) {
    _Float16* base = wT + (size_t)l * 655360;
    k_wt<<<192, 256, 0, stream>>>(sa_w + l * 49152, 128, 384, base, 49152);
    k_wt<<<64, 256, 0, stream>>>(sa_ow + l * 16384, 128, 128, base + 49152, 16384);
    k_wt<<<192, 256, 0, stream>>>(ca_w + l * 49152, 128, 384, base + 65536, 49152);
    k_wt<<<64, 256, 0, stream>>>(ca_ow + l * 16384, 128, 128, base + 114688, 16384);
    k_wt<<<1024, 256, 0, stream>>>(ff1_w + l * 262144, 128, 2048, base + 131072, 262144);
    k_wt<<<1024, 256, 0, stream>>>(ff2_w + l * 262144, 2048, 128, base + 393216, 262144);
  }
  k_wt<<<64, 256, 0, stream>>>(s1_w, 128, 128, WpT, 16384);
  k_wt<<<64, 256, 0, stream>>>(s1_w + 16384, 128, 128, WcT, 16384);
  k_wt<<<64, 256, 0, stream>>>(s1_w + 32768, 128, 128, WaT, 16384);

  // ---- encode prev (t=0) and curr (t=1) ----
  k_encode<<<1024, 128, 0, stream>>>(backbone, prev_info, 0, 0,
      coord_w, coord_b, roi_w, roi_b, roi_ln_g, roi_ln_b,
      fuse_w, fuse_b, fuse_ln_g, fuse_ln_b, type_emb, prev_tok);
  k_encode<<<1024, 128, 0, stream>>>(backbone, curr_info, 1, 1,
      coord_w, coord_b, roi_w, roi_b, roi_ln_g, roi_ln_b,
      fuse_w, fuse_b, fuse_ln_g, fuse_ln_b, type_emb, x);

  for (int l = 0; l < 2; ++l) {
    _Float16* base = wT + (size_t)l * 655360;
    // self-attention
    k_gemm<<<dim3(64, 3), 256, 0, stream>>>(x, 128, base, sa_b + l * 384,
        qkv, 384, qkv16, 384, 128, 0);
    k_attn<<<16, 256, 0, stream>>>(qkv16, 384, 0, qkv16, 384, 128, 256, attn, 256);
    k_gemm<<<dim3(64, 1), 256, 0, stream>>>(attn, 128, base + 49152, sa_ob + l * 128,
        sub, 128, nullptr, 0, 128, 0);
    k_resln<<<1024, 128, 0, stream>>>(x, sub, ln1_g + l * 128, ln1_b + l * 128);
    // cross-attention: q from x (cols 0..127), k/v from prev_tok (cols 128..383)
    k_gemm<<<dim3(64, 1), 256, 0, stream>>>(x, 128, base + 65536, ca_b + l * 384,
        qkv, 384, qkv16, 384, 128, 0);
    k_gemm<<<dim3(64, 2), 256, 0, stream>>>(prev_tok, 128, base + 65536 + (size_t)128 * 128,
        ca_b + l * 384 + 128, kvb + 128, 384, kv16 + 128, 384, 128, 0);
    k_attn<<<16, 256, 0, stream>>>(qkv16, 384, 0, kv16, 384, 128, 256, attn, 256);
    k_gemm<<<dim3(64, 1), 256, 0, stream>>>(attn, 128, base + 114688, ca_ob + l * 128,
        sub, 128, nullptr, 0, 128, 0);
    k_resln<<<1024, 128, 0, stream>>>(x, sub, ln2_g + l * 128, ln2_b + l * 128);
    // FFN
    k_gemm<<<dim3(64, 16), 256, 0, stream>>>(x, 128, base + 131072, ff1_b + l * 2048,
        ffb, 2048, nullptr, 0, 128, 1);
    k_gemm<<<dim3(64, 1), 256, 0, stream>>>(ffb, 2048, base + 393216, ff2_b + l * 128,
        sub, 128, nullptr, 0, 2048, 0);
    k_resln<<<1024, 128, 0, stream>>>(x, sub, ln3_g + l * 128, ln3_b + l * 128);
  }

  // scorer pre-projections
  k_gemm<<<dim3(64, 1), 256, 0, stream>>>(prev_tok, 128, WpT, nullptr,
      prevWp, 128, nullptr, 0, 128, 0);
  k_gemm<<<dim3(64, 1), 256, 0, stream>>>(x, 128, WcT, nullptr,
      hsWc, 128, nullptr, 0, 128, 0);

  k_fill<<<(4 * 257 * 257 + 255) / 256, 256, 0, stream>>>(aug, 4 * 257 * 257, -10000.0f);
  k_scorer<<<16384, 256, 0, stream>>>(x, prev_tok, prevWp, hsWc, WaT,
      s1_w, s1_b, s2_w, s2_b, prev_info, curr_info, aug);
  k_db<<<4, 256, 0, stream>>>(prev_tok, rdb_w, rdb_b, aug, 1);
  k_db<<<4, 256, 0, stream>>>(x, cdb_w, cdb_b, aug, 0);
}
